// GlobalNode_72387378807010
// MI455X (gfx1250) — compile-verified, matrix operand order checked by reading, not run on hardware
//
#include <hip/hip_runtime.h>

#define NN 100000
#define EE 300000
#define GG 4096
#define DD 256
#define LL 5
#define BN_EPS 1e-5f

typedef __attribute__((ext_vector_type(16))) __bf16 v16bf;
typedef __attribute__((ext_vector_type(8)))  float  v8f;

static __device__ __forceinline__ unsigned short f2bf_bits(float f) {
  unsigned u = __builtin_bit_cast(unsigned, f);
  unsigned r = u + 0x7FFFu + ((u >> 16) & 1u);   // round-to-nearest-even
  return (unsigned short)(r >> 16);
}
static __device__ __forceinline__ __bf16 bits2bf(unsigned short h) {
  return __builtin_bit_cast(__bf16, h);
}

// ---------------- encoders ----------------
__global__ __launch_bounds__(256) void encode_atoms(const int* __restrict__ x,
                                                    const float* __restrict__ emb,
                                                    float* __restrict__ h0) {
  int n = blockIdx.x, c = threadIdx.x;
  float s = 0.f;
#pragma unroll
  for (int f = 0; f < 9; ++f) {
    int idx = x[n * 9 + f];
    s += emb[((size_t)f * 120 + idx) * DD + c];
  }
  h0[(size_t)n * DD + c] = s;
}

__global__ __launch_bounds__(256) void init_vn(const float* __restrict__ vn_emb,
                                               float* __restrict__ vncur) {
  vncur[(size_t)blockIdx.x * DD + threadIdx.x] = vn_emb[threadIdx.x];
}

// ---------------- elementwise / graph ops ----------------
__global__ __launch_bounds__(256) void add_vn(const float* __restrict__ h,
                                              const float* __restrict__ vn,
                                              const int* __restrict__ batch,
                                              float* __restrict__ out) {
  int n = blockIdx.x, c = threadIdx.x;
  out[(size_t)n * DD + c] = h[(size_t)n * DD + c] + vn[(size_t)batch[n] * DD + c];
}

__global__ __launch_bounds__(256) void init_z(const float* __restrict__ hin,
                                              const float* __restrict__ eps,
                                              float* __restrict__ z) {
  int n = blockIdx.x, c = threadIdx.x;
  z[(size_t)n * DD + c] = (1.0f + eps[0]) * hin[(size_t)n * DD + c];
}

__global__ __launch_bounds__(256) void edge_msg(const int* __restrict__ src,
                                                const int* __restrict__ dst,
                                                const int* __restrict__ eattr,
                                                const float* __restrict__ bond_emb,
                                                const float* __restrict__ hin,
                                                float* __restrict__ z) {
  int e = blockIdx.x, c = threadIdx.x;
  int s = src[e], d = dst[e];
  float m = hin[(size_t)s * DD + c];
#pragma unroll
  for (int f = 0; f < 3; ++f) {
    int idx = eattr[e * 3 + f];
    m += bond_emb[((size_t)f * 6 + idx) * DD + c];
  }
  m = fmaxf(m, 0.f);
  atomicAdd(&z[(size_t)d * DD + c], m);
}

__global__ __launch_bounds__(256) void node_pool(const float* __restrict__ hin,
                                                 const int* __restrict__ batch,
                                                 float* __restrict__ pooled) {
  int n = blockIdx.x, c = threadIdx.x;
  atomicAdd(&pooled[(size_t)batch[n] * DD + c], hin[(size_t)n * DD + c]);
}

__global__ __launch_bounds__(256) void copybuf(const float* __restrict__ a,
                                               float* __restrict__ b) {
  size_t i = (size_t)blockIdx.x * DD + threadIdx.x;
  b[i] = a[i];
}

__global__ __launch_bounds__(256) void zerobuf(float* __restrict__ p, int n) {
  int i = blockIdx.x * 256 + threadIdx.x;
  if (i < n) p[i] = 0.f;
}

// ---------------- batchnorm ----------------
__global__ __launch_bounds__(256) void bn_partial(const float* __restrict__ x,
                                                  float* __restrict__ sum,
                                                  float* __restrict__ sumsq, int M) {
  int c = threadIdx.x;
  int r0 = blockIdx.x * 64;
  int r1 = r0 + 64; if (r1 > M) r1 = M;
  float s = 0.f, q = 0.f;
  for (int r = r0; r < r1; ++r) {
    float v = x[(size_t)r * DD + c];
    s += v; q += v * v;
  }
  atomicAdd(&sum[c], s);
  atomicAdd(&sumsq[c], q);
}

__global__ __launch_bounds__(256) void bn_finalize(const float* __restrict__ sum,
                                                   const float* __restrict__ sumsq,
                                                   const float* __restrict__ g,
                                                   const float* __restrict__ b,
                                                   float* __restrict__ scale,
                                                   float* __restrict__ shift, int M) {
  int c = threadIdx.x;
  float inv = 1.0f / (float)M;
  float mean = sum[c] * inv;
  float var  = sumsq[c] * inv - mean * mean;
  float sc   = g[c] * rsqrtf(var + BN_EPS);
  scale[c] = sc;
  shift[c] = b[c] - mean * sc;
}

__global__ __launch_bounds__(256) void bn_apply(const float* __restrict__ x,
                                                const float* __restrict__ scale,
                                                const float* __restrict__ shift,
                                                float* __restrict__ y, int relu) {
  int n = blockIdx.x, c = threadIdx.x;
  float v = x[(size_t)n * DD + c] * scale[c] + shift[c];
  if (relu) v = fmaxf(v, 0.f);
  y[(size_t)n * DD + c] = v;
}

// ---------------- weight pre-pack: fp32 W[256,256] -> bf16 fragments ----------------
// Wp layout: [ntile(16)][kkstep(8)][lane(32)][elem(16)] so a lane's whole B
// fragment for one wmma is 32 contiguous bytes (2x global_load_b128).
// Fragment element mapping (16-bit B 32x16, column per lane):
//   lanes 0-15:  col n=lane,   elems 0..7 -> K=kk+0..7,  elems 8..15 -> K=kk+16..23
//   lanes 16-31: col n=lane-16, elems 0..7 -> K=kk+8..15, elems 8..15 -> K=kk+24..31
__global__ __launch_bounds__(256) void pack_W(const float* __restrict__ W,
                                              __bf16* __restrict__ Wp) {
  int b = blockIdx.x;               // 128 blocks: [ntile(16) x kkstep(8)]
  int ntile = b >> 3, kks = b & 7;
  for (int i = threadIdx.x; i < 32 * 16; i += 256) {
    int lane = i >> 4, j = i & 15;
    int n = ntile * 16 + (lane & 15);
    int kb = (lane >> 4) << 3;
    int k = kks * 32 + ((j < 8) ? (kb + j) : (16 + kb + (j - 8)));
    Wp[(((size_t)ntile * 8 + kks) * 32 + lane) * 16 + j] =
        bits2bf(f2bf_bits(W[(size_t)k * DD + n]));
  }
}

// ---------------- WMMA GEMM: C[M,256] = A[M,256] @ W[256,256] + bias ----------------
// M is always a multiple of 16 here (100000, 4096) -> no bounds checks, no
// divergence anywhere (EXEC all-ones at every WMMA). Block = 256 threads =
// 8 waves; block owns a 16-row strip. A tile staged to LDS in fragment order
// (one 32B ds read per fragment); B fragments come pre-packed from Wp.
// 8 k-steps x 2 column tiles = 16 v_wmma_f32_16x16x32_bf16 per wave.
__global__ __launch_bounds__(256) void gemm_bias_wmma(const float* __restrict__ A,
                                                      const __bf16* __restrict__ Wp,
                                                      const float* __restrict__ bias,
                                                      float* __restrict__ C) {
  __shared__ __bf16 ldsA[8 * 32 * 16];   // [kkstep][lane][elem], 8 KB
  const int rowBase = blockIdx.x * 16;
  const int tid = threadIdx.x;

  // stage A tile (16x256 fp32 -> bf16) directly in fragment order
  for (int i = tid; i < 8 * 32 * 16; i += 256) {
    int kks  = i >> 9;
    int lane = (i >> 4) & 31;
    int j    = i & 15;
    int m    = lane & 15;
    int kb   = (lane >> 4) << 3;
    int k    = kks * 32 + ((j < 8) ? (kb + j) : (16 + kb + (j - 8)));
    ldsA[i] = bits2bf(f2bf_bits(A[(size_t)(rowBase + m) * DD + k]));
  }
  __syncthreads();

  const int wave = tid >> 5;
  const int lane = tid & 31;

#pragma unroll
  for (int t = 0; t < 2; ++t) {
    const int ntile = wave * 2 + t;
    const int n = ntile * 16 + (lane & 15);
    const __bf16* wp = Wp + (((size_t)ntile * 8) * 32 + lane) * 16;
    v8f acc = {};
#pragma unroll
    for (int kks = 0; kks < 8; ++kks) {
      v16bf a = *(const v16bf*)&ldsA[(kks * 32 + lane) * 16];
      v16bf bfr = *(const v16bf*)(wp + (size_t)kks * 32 * 16);
      acc = __builtin_amdgcn_wmma_f32_16x16x32_bf16(false, a, false, bfr,
                                                    (short)0, acc, false, false);
    }
    const float bv = bias[n];
    const int mbase = rowBase + ((lane >> 4) << 3);  // C layout: VGPR r -> M=r / M=r+8
#pragma unroll
    for (int r = 0; r < 8; ++r)
      C[(size_t)(mbase + r) * DD + n] = acc[r] + bv;
  }
}

// ---------------- host orchestration ----------------
extern "C" void kernel_launch(void* const* d_in, const int* in_sizes, int n_in,
                              void* d_out, int out_size, void* d_ws, size_t ws_size,
                              hipStream_t stream) {
  const int*   x          = (const int*)d_in[0];
  const int*   eidx       = (const int*)d_in[1];
  const int*   eattr      = (const int*)d_in[2];
  const int*   batch      = (const int*)d_in[3];
  const float* atom_emb   = (const float*)d_in[4];
  const float* bond_emb   = (const float*)d_in[5];
  const float* vn_emb     = (const float*)d_in[6];
  const float* gin_eps    = (const float*)d_in[7];
  const float* conv_W1    = (const float*)d_in[8];
  const float* conv_b1    = (const float*)d_in[9];
  const float* conv_bn_g  = (const float*)d_in[10];
  const float* conv_bn_b  = (const float*)d_in[11];
  const float* conv_W2    = (const float*)d_in[12];
  const float* conv_b2    = (const float*)d_in[13];
  const float* outer_bn_g = (const float*)d_in[14];
  const float* outer_bn_b = (const float*)d_in[15];
  const float* vn_W1      = (const float*)d_in[16];
  const float* vn_b1      = (const float*)d_in[17];
  const float* vn_bn1_g   = (const float*)d_in[18];
  const float* vn_bn1_b   = (const float*)d_in[19];
  const float* vn_W2      = (const float*)d_in[20];
  const float* vn_b2      = (const float*)d_in[21];
  const float* vn_bn2_g   = (const float*)d_in[22];
  const float* vn_bn2_b   = (const float*)d_in[23];

  const int* src = eidx;
  const int* dst = eidx + EE;

  float* out = (float*)d_out;                       // [L+1, N, D]
  float* ws  = (float*)d_ws;
  float* z      = ws;                               // N*D   (GIN input / z2 / carried h)
  float* t1     = z + (size_t)NN * DD;              // N*D   (mid-MLP activations)
  float* pooled = t1 + (size_t)NN * DD;             // G*D
  float* vncur  = pooled + (size_t)GG * DD;         // G*D
  float* vntmp  = vncur + (size_t)GG * DD;          // G*D
  float* ssum   = vntmp + (size_t)GG * DD;          // D
  float* ssq    = ssum + DD;                        // D (contiguous with ssum)
  float* scale  = ssq + DD;                         // D
  float* shift  = scale + DD;                       // D
  __bf16* Wp    = (__bf16*)(shift + DD);            // 256*256 bf16 = 128 KB

  auto bn = [&](const float* xb, float* yb, const float* g, const float* b,
                int M, int relu) {
    zerobuf<<<2, 256, 0, stream>>>(ssum, 2 * DD);   // zero ssum+ssq every call
    bn_partial<<<(M + 63) / 64, 256, 0, stream>>>(xb, ssum, ssq, M);
    bn_finalize<<<1, DD, 0, stream>>>(ssum, ssq, g, b, scale, shift, M);
    bn_apply<<<M, 256, 0, stream>>>(xb, scale, shift, yb, relu);
  };

  auto gemm = [&](const float* A, const float* W, const float* bias,
                  float* C, int M) {
    pack_W<<<128, 256, 0, stream>>>(W, Wp);
    gemm_bias_wmma<<<M / 16, 256, 0, stream>>>(A, Wp, bias, C);
  };

  encode_atoms<<<NN, 256, 0, stream>>>(x, atom_emb, out);   // slot 0 = h0
  init_vn<<<GG, 256, 0, stream>>>(vn_emb, vncur);

  for (int l = 0; l < LL; ++l) {
    float* hin = out + (size_t)l * NN * DD;         // reference mutates h_list[l] = h_in
    const float* hsrc = (l == 0) ? hin : z;         // z carries previous layer's output
    add_vn<<<NN, 256, 0, stream>>>(hsrc, vncur, batch, hin);

    // z = (1+eps)*h_in, then scatter-add relu(h_in[src] + bond_enc(e)) at dst
    init_z<<<NN, 256, 0, stream>>>(hin, gin_eps + l, z);
    edge_msg<<<EE, 256, 0, stream>>>(src, dst, eattr, bond_emb, hin, z);

    // GIN MLP: Lin -> BN -> ReLU -> Lin
    gemm(z, conv_W1 + (size_t)l * DD * DD, conv_b1 + l * DD, t1, NN);
    bn(t1, t1, conv_bn_g + l * DD, conv_bn_b + l * DD, NN, 1);
    gemm(t1, conv_W2 + (size_t)l * DD * DD, conv_b2 + l * DD, z, NN);

    if (l < LL - 1) {
      bn(z, z, outer_bn_g + l * DD, outer_bn_b + l * DD, NN, 1);  // carried in z

      // virtual-node update: pooled = segsum(h_in) + vn; two-layer MLP with BNs
      copybuf<<<GG, 256, 0, stream>>>(vncur, pooled);
      node_pool<<<NN, 256, 0, stream>>>(hin, batch, pooled);
      gemm(pooled, vn_W1 + (size_t)l * DD * DD, vn_b1 + l * DD, vntmp, GG);
      bn(vntmp, vntmp, vn_bn1_g + l * DD, vn_bn1_b + l * DD, GG, 1);
      gemm(vntmp, vn_W2 + (size_t)l * DD * DD, vn_b2 + l * DD, pooled, GG);
      bn(pooled, vncur, vn_bn2_g + l * DD, vn_bn2_b + l * DD, GG, 1);
    } else {
      // last layer: outer BN, no relu, into output slot L
      bn(z, out + (size_t)LL * NN * DD, outer_bn_g + l * DD, outer_bn_b + l * DD, NN, 0);
    }
  }
}
